// SpatialRelationGATLayer_69166153335440
// MI455X (gfx1250) — matrix operand order repaired and checked by reading, not compiled, not run on hardware
//
#include <hip/hip_runtime.h>
#include <hip/hip_bf16.h>

// Problem constants
#define BQ  32
#define NN  256
#define PP  64
#define TT  768
#define HH  8
#define EE  512
#define EDD 768
#define GDD 64
#define DD  96

typedef __attribute__((ext_vector_type(16))) __bf16 v16bf;
typedef __attribute__((ext_vector_type(8)))  float  v8f;

__device__ __forceinline__ unsigned short f2bf(float x) {
  unsigned int u = __float_as_uint(x);
  u += 0x7FFFu + ((u >> 16) & 1u);   // round-to-nearest-even
  return (unsigned short)(u >> 16);
}

union FragU {
  uint4 q[2];
  v16bf v;
};

// A fragment: 16x32 bf16, row-major source [*, ld], rows row0..row0+15, cols k0..k0+31.
// ISA layout: lanes 0-15 hold M=lane, K in {0..7, 16..23}; lanes 16-31 hold K in {8..15, 24..31}.
__device__ __forceinline__ v16bf load_frag_a(const unsigned short* A, int ld, int row0, int k0) {
  int lane = threadIdx.x & 31;
  const unsigned short* p = A + (size_t)(row0 + (lane & 15)) * ld + k0 + (lane >> 4) * 8;
  FragU f;
  f.q[0] = *reinterpret_cast<const uint4*>(p);        // k .. k+7
  f.q[1] = *reinterpret_cast<const uint4*>(p + 16);   // k+16 .. k+23
  return f.v;
}

// B fragment: 32x16 bf16 from packed [Ncol, K] storage (column n is a contiguous K-row).
// Lanes 0-15 hold K in {0..7, 16..23} of column n=lane; lanes 16-31 hold K in {8..15, 24..31}.
__device__ __forceinline__ v16bf load_frag_b(const unsigned short* B, int ld, int col0, int k0) {
  int lane = threadIdx.x & 31;
  const unsigned short* p = B + (size_t)(col0 + (lane & 15)) * ld + k0 + (lane >> 4) * 8;
  FragU f;
  f.q[0] = *reinterpret_cast<const uint4*>(p);
  f.q[1] = *reinterpret_cast<const uint4*>(p + 16);
  return f.v;
}

// Register-blocked GEMM: one wave computes a (RT*16) x (CT*16) f32 tile.
// RT A-fragments + CT B-fragments per 32-deep k-step feed RT*CT WMMAs
// (32x64 tile -> 12 b128 loads per 8 v_wmma = 1.5 vmem/wmma).
// K-loop unroll capped at 2 so live VGPRs (accums + 2 iterations of fragments)
// stay below 256 and the backend does not spill the C tiles to scratch.
// C/D layout: VGPR i holds row +i+8*(lane>=16), col = lane&15.
template <int RT, int CT, typename F>
__device__ __forceinline__ void run_gemm(const unsigned short* A, int lda,
                                         const unsigned short* B, int ldb,
                                         int row0, int col0, int K, F&& epi) {
  v8f acc[RT][CT];
#pragma unroll
  for (int r = 0; r < RT; ++r)
#pragma unroll
    for (int c = 0; c < CT; ++c)
      acc[r][c] = v8f{0.f,0.f,0.f,0.f,0.f,0.f,0.f,0.f};

#pragma unroll 2
  for (int k0 = 0; k0 < K; k0 += 32) {
    v16bf af[RT], bfr[CT];
#pragma unroll
    for (int r = 0; r < RT; ++r) af[r] = load_frag_a(A, lda, row0 + 16 * r, k0);
#pragma unroll
    for (int c = 0; c < CT; ++c) bfr[c] = load_frag_b(B, ldb, col0 + 16 * c, k0);
#pragma unroll
    for (int r = 0; r < RT; ++r)
#pragma unroll
      for (int c = 0; c < CT; ++c)
        acc[r][c] = __builtin_amdgcn_wmma_f32_16x16x32_bf16(
            false, af[r], false, bfr[c], (short)0, acc[r][c], false, false);
  }

  int lane = threadIdx.x & 31;
  int cl = lane & 15, rb = (lane >> 4) * 8;
#pragma unroll
  for (int r = 0; r < RT; ++r)
#pragma unroll
    for (int c = 0; c < CT; ++c)
#pragma unroll
      for (int i = 0; i < 8; ++i)
        epi(row0 + 16 * r + rb + i, col0 + 16 * c + cl, acc[r][c][i]);
}

__device__ __forceinline__ int wave_id() {
  return blockIdx.x * (blockDim.x >> 5) + (threadIdx.x >> 5);
}

// ---------------- weight pack: f32 [K, Nc] -> bf16 transposed [Nc, K] ----------------
__global__ void k_conv_t(const float* __restrict__ src, unsigned short* __restrict__ dst,
                         int K, int Nc) {
  int idx = blockIdx.x * blockDim.x + threadIdx.x;
  if (idx >= K * Nc) return;
  int c = idx / K, k = idx - c * K;
  dst[idx] = f2bf(src[(size_t)k * Nc + c]);
}

// ---------------- LN over P=64, output bf16 ----------------
__global__ void k_lnx(const float* __restrict__ x, const float* __restrict__ g,
                      const float* __restrict__ b, unsigned short* __restrict__ out, int rows) {
  int w = wave_id();
  if (w >= rows) return;
  int lane = threadIdx.x & 31;
  const float* xp = x + (size_t)w * 64;
  float x0 = xp[lane], x1 = xp[lane + 32];
  float s = x0 + x1, sq = x0 * x0 + x1 * x1;
  for (int m = 16; m; m >>= 1) { s += __shfl_xor(s, m); sq += __shfl_xor(sq, m); }
  float mean = s * (1.f / 64.f);
  float var  = sq * (1.f / 64.f) - mean * mean;
  float rs   = rsqrtf(var + 1e-5f);
  unsigned short* op = out + (size_t)w * 64;
  op[lane]      = f2bf((x0 - mean) * rs * g[lane]      + b[lane]);
  op[lane + 32] = f2bf((x1 - mean) * rs * g[lane + 32] + b[lane + 32]);
}

// ---------------- GEMM [8192,K]x[K,768] + bias + SiLU -> bf16 (32x64 tiles) ----------------
__global__ void __launch_bounds__(128, 1)
k_gemm_silu(const unsigned short* __restrict__ A, const unsigned short* __restrict__ Bp,
            const float* __restrict__ bias, unsigned short* __restrict__ out, int K) {
  const int M = BQ * NN, Nc = TT, tn = Nc >> 6;  // 12
  int w = wave_id();
  if (w >= (M >> 5) * tn) return;
  int row0 = (w / tn) << 5, col0 = (w % tn) << 6;
  run_gemm<2, 4>(A, K, Bp, K, row0, col0, K, [&](int R, int C, float v) {
    v += bias[C];
    out[(size_t)R * Nc + C] = f2bf(v / (1.f + __expf(-v)));
  });
}

// ---------------- GEMM + bias + node-mask -> bf16 ----------------
__global__ void __launch_bounds__(128, 1)
k_gemm_mask(const unsigned short* __restrict__ A, const unsigned short* __restrict__ Bp,
            const float* __restrict__ bias, const float* __restrict__ nm,
            unsigned short* __restrict__ out, int K) {
  const int M = BQ * NN, Nc = TT, tn = Nc >> 6;
  int w = wave_id();
  if (w >= (M >> 5) * tn) return;
  int row0 = (w / tn) << 5, col0 = (w % tn) << 6;
  run_gemm<2, 4>(A, K, Bp, K, row0, col0, K, [&](int R, int C, float v) {
    out[(size_t)R * Nc + C] = f2bf((v + bias[C]) * nm[R]);
  });
}

// ---------------- Q/K/V projection; scatter to [B,H,N,D] (or [B,H,D,N] for V) bf16 ----------
__global__ void __launch_bounds__(128, 1)
k_gemm_qkv(const unsigned short* __restrict__ A, const unsigned short* __restrict__ Bp,
           const float* __restrict__ bias, unsigned short* __restrict__ dst,
           int transposed) {
  const int M = BQ * NN, K = TT, Nc = TT, tn = Nc >> 6;
  int w = wave_id();
  if (w >= (M >> 5) * tn) return;
  int row0 = (w / tn) << 5, col0 = (w % tn) << 6;
  run_gemm<2, 4>(A, K, Bp, K, row0, col0, K, [&](int R, int C, float v) {
    v += bias[C];
    int b = R >> 8, n = R & 255;
    int h = C / DD, d = C - h * DD;
    size_t o = transposed ? ((size_t)((b * HH + h) * DD + d) * NN + n)
                          : ((size_t)((b * HH + h) * NN + n) * DD + d);
    dst[o] = f2bf(v);
  });
}

// ---------------- score buffer init: diag(active nodes)=0, else -BIG sentinel ----------------
__global__ void k_init_scores(float* __restrict__ sc, const float* __restrict__ nm) {
  size_t idx = (size_t)blockIdx.x * blockDim.x + threadIdx.x;
  if (idx >= (size_t)BQ * HH * NN * NN) return;
  int j  = (int)(idx & 255);
  int i  = (int)((idx >> 8) & 255);
  int bh = (int)(idx >> 16);
  int b  = bh >> 3;
  sc[idx] = (i == j && nm[b * NN + i] > 0.5f) ? 0.f : -3.0e38f;
}

// ---------------- edge bias: one wave per (b,e); writes symmetric entries ----------------
__global__ void k_edge(const int* __restrict__ edges, const float* __restrict__ rmask,
                       const float* __restrict__ remb, const float* __restrict__ geo,
                       const float* __restrict__ lg, const float* __restrict__ lb,
                       const float* __restrict__ wg, const float* __restrict__ bg,
                       const float* __restrict__ we, const float* __restrict__ be,
                       const float* __restrict__ nm, float* __restrict__ sc) {
  int w = wave_id();
  if (w >= BQ * EE) return;
  int lane = threadIdx.x & 31;
  int b = w / EE;
  // LN of geo features (64)
  const float* gp = geo + (size_t)w * GDD;
  float x0 = gp[lane], x1 = gp[lane + 32];
  float s = x0 + x1, sq = x0 * x0 + x1 * x1;
  for (int m = 16; m; m >>= 1) { s += __shfl_xor(s, m); sq += __shfl_xor(sq, m); }
  float mean = s * (1.f / 64.f);
  float var  = sq * (1.f / 64.f) - mean * mean;
  float rs   = rsqrtf(var + 1e-5f);
  float y0 = (x0 - mean) * rs * lg[lane]      + lb[lane];
  float y1 = (x1 - mean) * rs * lg[lane + 32] + lb[lane + 32];
  float acc[HH];
#pragma unroll
  for (int h = 0; h < HH; ++h)
    acc[h] = y0 * wg[lane * HH + h] + y1 * wg[(lane + 32) * HH + h];
  // text path: remb[768] @ we[768,8]
  const float* rp = remb + (size_t)w * EDD;
  for (int kk = lane; kk < EDD; kk += 32) {
    float r = rp[kk];
#pragma unroll
    for (int h = 0; h < HH; ++h) acc[h] += r * we[kk * HH + h];
  }
#pragma unroll
  for (int h = 0; h < HH; ++h)
    for (int m = 16; m; m >>= 1) acc[h] += __shfl_xor(acc[h], m);

  int src = edges[(size_t)w * 2], dst = edges[(size_t)w * 2 + 1];
  src = min(max(src, 0), NN - 1);
  dst = min(max(dst, 0), NN - 1);
  bool ok = (rmask[w] > 0.5f) && (nm[b * NN + src] > 0.5f) && (nm[b * NN + dst] > 0.5f);
  if (ok && lane < 16) {
    int h = lane & 7;
    float v = acc[h] + be[h] + bg[h];
    size_t o = (lane < 8) ? ((size_t)((b * HH + h) * NN + src) * NN + dst)
                          : ((size_t)((b * HH + h) * NN + dst) * NN + src);
    sc[o] = v;
  }
}

// ---------------- QK^T per head, combine with bias/mask sentinel (32x64 tiles) ----------------
__global__ void __launch_bounds__(128, 1)
k_gemm_scores(const unsigned short* __restrict__ qb,
              const unsigned short* __restrict__ kb,
              float* __restrict__ sc) {
  const int tilesPerHead = (NN / 32) * (NN / 64);  // 32
  int w = wave_id();
  if (w >= BQ * HH * tilesPerHead) return;
  int head = w / tilesPerHead, t = w - head * tilesPerHead;
  const int tn = NN >> 6;  // 4
  int row0 = (t / tn) << 5, col0 = (t % tn) << 6;
  const unsigned short* A = qb + (size_t)head * NN * DD;
  const unsigned short* B = kb + (size_t)head * NN * DD;  // K rows are packed-B columns
  const float scale = 0.10206207261596575f;               // 1/sqrt(96)
  float* S = sc + (size_t)head * NN * NN;
  run_gemm<2, 4>(A, DD, B, DD, row0, col0, DD, [&](int R, int C, float v) {
    size_t o = (size_t)R * NN + C;
    float bv = S[o];
    S[o] = (bv < -1e37f) ? -3.0e38f : fmaf(v, scale, bv);
  });
}

// ---------------- softmax over rows of 256 -> bf16 probabilities ----------------
__global__ void k_softmax(const float* __restrict__ sc, unsigned short* __restrict__ p, int rows) {
  int w = wave_id();
  if (w >= rows) return;
  int lane = threadIdx.x & 31;
  const float* x = sc + (size_t)w * NN;
  float v[8], m = -3.4e38f;
#pragma unroll
  for (int t = 0; t < 8; ++t) { v[t] = x[lane + 32 * t]; m = fmaxf(m, v[t]); }
  for (int s = 16; s; s >>= 1) m = fmaxf(m, __shfl_xor(m, s));
  unsigned short* op = p + (size_t)w * NN;
  if (m < -1e37f) {  // fully masked row: final output gets zeroed by node mask anyway
#pragma unroll
    for (int t = 0; t < 8; ++t) op[lane + 32 * t] = 0;
    return;
  }
  float sum = 0.f;
#pragma unroll
  for (int t = 0; t < 8; ++t) { v[t] = __expf(v[t] - m); sum += v[t]; }
  for (int s = 16; s; s >>= 1) sum += __shfl_xor(sum, s);
  float inv = 1.f / sum;
#pragma unroll
  for (int t = 0; t < 8; ++t) op[lane + 32 * t] = f2bf(v[t] * inv);
}

// ---------------- P @ V per head -> attn_out bf16 [B,N,T] (32x32 tiles; 96 cols) ----------------
__global__ void __launch_bounds__(128, 1)
k_gemm_pv(const unsigned short* __restrict__ p,
          const unsigned short* __restrict__ vt,
          unsigned short* __restrict__ aout) {
  const int tilesPerHead = (NN / 32) * (DD / 32);  // 24
  int w = wave_id();
  if (w >= BQ * HH * tilesPerHead) return;
  int head = w / tilesPerHead, t = w - head * tilesPerHead;
  const int tn = DD >> 5;  // 3
  int row0 = (t / tn) << 5, col0 = (t % tn) << 5;
  const unsigned short* A = p  + (size_t)head * NN * NN;  // [256,256]
  const unsigned short* B = vt + (size_t)head * DD * NN;  // packed [96,256]
  int b = head >> 3, h = head & 7;
  run_gemm<2, 2>(A, NN, B, NN, row0, col0, NN, [&](int R, int C, float v) {
    aout[(size_t)(b * NN + R) * TT + h * DD + C] = f2bf(v);
  });
}

// ---------------- final projection -> f32 output ----------------
__global__ void __launch_bounds__(128, 1)
k_gemm_out(const unsigned short* __restrict__ A, const unsigned short* __restrict__ Bp,
           const float* __restrict__ bias, const float* __restrict__ nm,
           float* __restrict__ out) {
  const int M = BQ * NN, K = TT, Nc = TT, tn = Nc >> 6;
  int w = wave_id();
  if (w >= (M >> 5) * tn) return;
  int row0 = (w / tn) << 5, col0 = (w % tn) << 6;
  run_gemm<2, 4>(A, K, Bp, K, row0, col0, K, [&](int R, int C, float v) {
    out[(size_t)R * Nc + C] = (v + bias[C]) * nm[R];
  });
}

extern "C" void kernel_launch(void* const* d_in, const int* in_sizes, int n_in,
                              void* d_out, int out_size, void* d_ws, size_t ws_size,
                              hipStream_t stream) {
  (void)in_sizes; (void)n_in; (void)out_size; (void)ws_size;
  const float* xyxy  = (const float*)d_in[0];
  const float* nmask = (const float*)d_in[1];
  const int*   edges = (const int*)d_in[2];
  const float* rmask = (const float*)d_in[3];
  const float* remb  = (const float*)d_in[4];
  const float* geo   = (const float*)d_in[5];
  const float* ln1g  = (const float*)d_in[6];
  const float* ln1b  = (const float*)d_in[7];
  const float* w1    = (const float*)d_in[8];
  const float* b1    = (const float*)d_in[9];
  const float* w2    = (const float*)d_in[10];
  const float* b2    = (const float*)d_in[11];
  const float* wq    = (const float*)d_in[12];
  const float* bq    = (const float*)d_in[13];
  const float* wk    = (const float*)d_in[14];
  const float* bk    = (const float*)d_in[15];
  const float* wv    = (const float*)d_in[16];
  const float* bv    = (const float*)d_in[17];
  const float* we    = (const float*)d_in[18];
  const float* be    = (const float*)d_in[19];
  const float* lngg  = (const float*)d_in[20];
  const float* lngb  = (const float*)d_in[21];
  const float* wg    = (const float*)d_in[22];
  const float* bg    = (const float*)d_in[23];
  const float* wo    = (const float*)d_in[24];
  const float* bo    = (const float*)d_in[25];
  float* out = (float*)d_out;

  // workspace carve-up (256B aligned regions)
  char* ws = (char*)d_ws;
  size_t off = 0;
  auto carve = [&](size_t elems, size_t esz) -> char* {
    char* p = ws + off;
    off += (elems * esz + 255) & ~(size_t)255;
    return p;
  };
  unsigned short* w1p  = (unsigned short*)carve((size_t)TT * PP, 2);
  unsigned short* w2p  = (unsigned short*)carve((size_t)TT * TT, 2);
  unsigned short* wqp  = (unsigned short*)carve((size_t)TT * TT, 2);
  unsigned short* wkp  = (unsigned short*)carve((size_t)TT * TT, 2);
  unsigned short* wvp  = (unsigned short*)carve((size_t)TT * TT, 2);
  unsigned short* wop  = (unsigned short*)carve((size_t)TT * TT, 2);
  unsigned short* lnx  = (unsigned short*)carve((size_t)BQ * NN * PP, 2);
  unsigned short* h1   = (unsigned short*)carve((size_t)BQ * NN * TT, 2);
  unsigned short* sbuf = (unsigned short*)carve((size_t)BQ * NN * TT, 2);
  unsigned short* qb   = (unsigned short*)carve((size_t)BQ * HH * NN * DD, 2);
  unsigned short* kb   = (unsigned short*)carve((size_t)BQ * HH * NN * DD, 2);
  unsigned short* vt   = (unsigned short*)carve((size_t)BQ * HH * DD * NN, 2);
  float*          scb  = (float*)carve((size_t)BQ * HH * NN * NN, 4);
  unsigned short* pb   = (unsigned short*)carve((size_t)BQ * HH * NN * NN, 2);
  unsigned short* aout = (unsigned short*)carve((size_t)BQ * NN * TT, 2);

  // 1) pack weights to bf16, transposed [Ncol, K]
  k_conv_t<<<(PP * TT + 255) / 256, 256, 0, stream>>>(w1, w1p, PP, TT);
  k_conv_t<<<(TT * TT + 255) / 256, 256, 0, stream>>>(w2, w2p, TT, TT);
  k_conv_t<<<(TT * TT + 255) / 256, 256, 0, stream>>>(wq, wqp, TT, TT);
  k_conv_t<<<(TT * TT + 255) / 256, 256, 0, stream>>>(wk, wkp, TT, TT);
  k_conv_t<<<(TT * TT + 255) / 256, 256, 0, stream>>>(wv, wvp, TT, TT);
  k_conv_t<<<(TT * TT + 255) / 256, 256, 0, stream>>>(wo, wop, TT, TT);
  // 2) LN of node embeddings (8192 rows, 8 waves/block)
  k_lnx<<<1024, 256, 0, stream>>>(xyxy, ln1g, ln1b, lnx, BQ * NN);
  // 3) MLP: 64->768 + SiLU, then 768->768 + mask   (3072 waves each, 32x64 tiles)
  k_gemm_silu<<<768, 128, 0, stream>>>(lnx, w1p, b1, h1, PP);
  k_gemm_mask<<<768, 128, 0, stream>>>(h1, w2p, b2, nmask, sbuf, TT);
  // 4) Q/K/V projections with head scatter (V transposed for PV GEMM)
  k_gemm_qkv<<<768, 128, 0, stream>>>(sbuf, wqp, bq, qb, 0);
  k_gemm_qkv<<<768, 128, 0, stream>>>(sbuf, wkp, bk, kb, 0);
  k_gemm_qkv<<<768, 128, 0, stream>>>(sbuf, wvp, bv, vt, 1);
  // 5) score buffer: mask sentinel + self-loop diagonal, then edge-bias scatter
  k_init_scores<<<65536, 256, 0, stream>>>(scb, nmask);
  k_edge<<<2048, 256, 0, stream>>>(edges, rmask, remb, geo, lngg, lngb, wg, bg, we, be, nmask, scb);
  // 6) QK^T + bias/mask combine   (8192 waves)
  k_gemm_scores<<<2048, 128, 0, stream>>>(qb, kb, scb);
  // 7) softmax -> bf16 probabilities (65536 rows, 8 waves/block)
  k_softmax<<<8192, 256, 0, stream>>>(scb, pb, BQ * HH * NN);
  // 8) P @ V -> attn output [B,N,T]   (6144 waves)
  k_gemm_pv<<<1536, 128, 0, stream>>>(pb, vt, aout);
  // 9) output projection + bias + node mask -> f32
  k_gemm_out<<<768, 128, 0, stream>>>(aout, wop, bo, nmask, out);
}
